// NMP_Duvenaud_67740224192591
// MI455X (gfx1250) — compile-verified, matrix operand order checked by reading, not run on hardware
//
#include <hip/hip_runtime.h>
#include <math.h>

// ---------- types ----------
typedef __attribute__((ext_vector_type(16))) __bf16 v16bf;
typedef __attribute__((ext_vector_type(8)))  __bf16 v8bf;
typedef __attribute__((ext_vector_type(8)))  float  v8f;

__device__ __host__ inline __bf16 f2bf(float f) {
  union { float f; unsigned int u; } x; x.f = f;
  unsigned int r = (x.u + 0x7FFFu + ((x.u >> 16) & 1u)) >> 16;
  union { unsigned short s; __bf16 b; } y; y.s = (unsigned short)r;
  return y.b;
}
__device__ __host__ inline float bf2f(__bf16 b) {
  union { unsigned short s; __bf16 b; } y; y.b = b;
  union { float f; unsigned int u; } x; x.u = ((unsigned int)y.s) << 16;
  return x.f;
}

// ---------- WMMA fragment access (CDNA5 ISA 7.12.2 layouts, wave32) ----------
// A tile 16x32 bf16, row-major source, stride ld (ld multiple of 8, base 16B-aligned).
// lane m=lane&15 holds row m; half=lane>>4 offsets K by 8. Two 16B vector loads.
__device__ inline v16bf load_frag_a(const __bf16* base, int ld) {
  int lane = threadIdx.x & 31;
  int m = lane & 15, half = lane >> 4;
  const __bf16* row = base + m * ld + half * 8;
  v8bf lo = *(const v8bf*)(row);        // K = {0..7}   + 8*half
  v8bf hi = *(const v8bf*)(row + 16);   // K = {16..23} + 8*half
  return __builtin_shufflevector(lo, hi, 0, 1, 2, 3, 4, 5, 6, 7,
                                         8, 9, 10, 11, 12, 13, 14, 15);
}

// B operands are stored pre-swizzled: per (Kchunk, Ntile) block of 32x16=512 bf16,
// lane 16*half+n owns its 16 fragment values contiguously -> one 32B load.
__device__ inline v16bf load_frag_b_swz(const __bf16* blockbase) {
  int lane = threadIdx.x & 31;
  return *(const v16bf*)(blockbase + lane * 16);
}

// Swizzled index for B-matrix element (row=K-index, col=N-index), Ncols mult of 16.
__device__ inline int bswz_idx(int Ncols, int row, int col) {
  int c = row >> 5, kk = row & 31;
  int ct = col >> 4, n16 = col & 15;
  int half = (kk >> 3) & 1;
  int j = ((kk & 7) >> 1) + ((kk & 16) ? 4 : 0);
  int i = 2 * j + (kk & 1);
  return ((c * (Ncols >> 4) + ct) * 32 + half * 16 + n16) * 16 + i;
}

#define WMMA_BF16(A, B, C) \
  __builtin_amdgcn_wmma_f32_16x16x32_bf16(false, (A), false, (B), (short)0, (C), false, false)

// ---------- constants ----------
#define B_  128
#define N_  128
#define NODE_F 64
#define EDGE_F 32
#define OUT_F 128
#define RCAT 320   // 64 + 128 + 128
#define ROUT 256
#define TGT 12

// ---------- kernel 1: edge aggregation + degree buckets + bf16 converts ----------
// One wave per (b,v). Exploits adjacency sparsity (deg<=4 of 128): the e-row
// load is skipped under a wave-uniform branch when g[v,w]==0, cutting e traffic
// from 268MB to ~7MB. me is shared by BOTH layers (computed once).
// h0 is written row-major (for readout) AND B-swizzled (for layer-0 WMMA).
__global__ __launch_bounds__(256) void edge_prep_kernel(
    const float* __restrict__ g, const float* __restrict__ h_in,
    const float* __restrict__ e,
    __bf16* __restrict__ gbf, __bf16* __restrict__ h0bf,
    __bf16* __restrict__ h0swz,
    __bf16* __restrict__ mebf, int* __restrict__ kidx) {
  int gw = blockIdx.x * 8 + (threadIdx.x >> 5);
  int lane = threadIdx.x & 31;
  int b = gw >> 7, v = gw & 127;
  size_t row = (size_t)b * N_ + v;
  const float* grow = g + row * N_;
  const float* erow = e + row * N_ * EDGE_F;
  float acc = 0.f, deg = 0.f;
  for (int w = 0; w < N_; ++w) {
    float gv = grow[w];          // wave-uniform
    deg += gv;
    if (gv != 0.f) acc += gv * erow[w * EDGE_F + lane];  // uniform branch: skip ~97% of e
  }
  mebf[row * EDGE_F + lane] = f2bf(acc);
  for (int w = lane; w < N_; w += 32) gbf[row * N_ + w] = f2bf(grow[w]);
  const float* hrow = h_in + row * NODE_F;
  __bf16* hsw = h0swz + (size_t)b * N_ * NODE_F;
  for (int c = lane; c < NODE_F; c += 32) {
    __bf16 hv = f2bf(hrow[c]);
    h0bf[row * NODE_F + c] = hv;
    hsw[bswz_idx(NODE_F, v, c)] = hv;
  }
  if (lane == 0) {
    int d = (int)(deg + 0.5f);
    kidx[row] = (d >= 1 && d <= 4) ? (d - 1) : -1;   // DEGS = [1,2,3,4]
  }
}

// ---------- kernel 2: f32 -> bf16 convert into B-swizzled fragment layout ----------
__global__ void f2bf_swz_kernel(const float* __restrict__ src, __bf16* __restrict__ dst,
                                int rows, int ncols, int rowoff) {
  int idx = blockIdx.x * 256 + threadIdx.x;
  if (idx >= rows * ncols) return;
  int r = idx / ncols, c = idx % ncols;
  dst[bswz_idx(ncols, rowoff + r, c)] = f2bf(src[idx]);
}

// ---------- kernel 3: one GNN layer, all-WMMA ----------
// Per block: one batch b, 8 waves; wave owns 16-row strip (all deps wave-local,
// no __syncthreads in the hot path).
// Phase A: mh = g @ h  (bf16 WMMA, f32 acc) -> LDS message matrix m=[mh|me]
// Phase B: z_k = m @ W[k] for k=0..3 (WMMA), per-row bucket select + sigmoid.
template <int DH, int INF, bool WSWZ>
__global__ __launch_bounds__(256) void layer_kernel(
    const __bf16* __restrict__ gbf,    // [B,128,128] row-major (A operand)
    const __bf16* __restrict__ hswz,   // [B, N*DH] B-swizzled
    const __bf16* __restrict__ mebf,   // [B,128,32]
    const int*    __restrict__ kidx,   // [B,128]
    const __bf16* __restrict__ Wswz,   // [4, INF*128] B-swizzled per k
    __bf16* __restrict__ hout,         // [B,128,128] row-major
    __bf16* __restrict__ hout_swz) {   // [B, N*128] B-swizzled (if WSWZ)
  __shared__ __attribute__((aligned(32))) __bf16 ms[N_ * INF];
  const int b = blockIdx.x;
  const int wid = threadIdx.x >> 5;
  const int lane = threadIdx.x & 31;
  const int half = lane >> 4, n16 = lane & 15;
  const int row0 = wid * 16;

  const __bf16* gb = gbf + (size_t)b * N_ * N_;
  const __bf16* hb = hswz + (size_t)b * N_ * DH;
  const __bf16* me = mebf + (size_t)b * N_ * EDGE_F;

  // copy this strip's me columns into m
  for (int i = lane; i < 16 * EDGE_F; i += 32) {
    int r = i >> 5, c = i & 31;
    ms[(row0 + r) * INF + DH + c] = me[(row0 + r) * EDGE_F + c];
  }

  // Phase A: mh = g @ h for this strip
  for (int dt = 0; dt < DH / 16; ++dt) {
    v8f acc = {};
#pragma unroll
    for (int kc = 0; kc < N_ / 32; ++kc) {
      v16bf a = load_frag_a(gb + row0 * N_ + kc * 32, N_);
      v16bf bb = load_frag_b_swz(hb + (kc * (DH / 16) + dt) * 512);
      acc = WMMA_BF16(a, bb, acc);
    }
#pragma unroll
    for (int r = 0; r < 8; ++r)
      ms[(row0 + r + half * 8) * INF + dt * 16 + n16] = f2bf(acc[r]);
  }

  // Phase B: z_k = m @ W[k], bucket-select, sigmoid
  int krow[8];
  const int* kb = kidx + b * N_;
#pragma unroll
  for (int r = 0; r < 8; ++r) krow[r] = kb[row0 + r + half * 8];
  __bf16* od = hout + (size_t)b * N_ * OUT_F;
  __bf16* osw = WSWZ ? (hout_swz + (size_t)b * N_ * OUT_F) : (__bf16*)0;

  for (int nt = 0; nt < OUT_F / 16; ++nt) {
    float hsel[8];
#pragma unroll
    for (int r = 0; r < 8; ++r) hsel[r] = 0.f;
    for (int k = 0; k < 4; ++k) {
      v8f z = {};
#pragma unroll
      for (int c = 0; c < INF / 32; ++c) {
        v16bf a = load_frag_a(ms + row0 * INF + c * 32, INF);
        v16bf bb = load_frag_b_swz(Wswz + (size_t)k * INF * OUT_F +
                                   (c * (OUT_F / 16) + nt) * 512);
        z = WMMA_BF16(a, bb, z);
      }
#pragma unroll
      for (int r = 0; r < 8; ++r)
        if (krow[r] == k) hsel[r] = 1.f / (1.f + __expf(-z[r]));
    }
#pragma unroll
    for (int r = 0; r < 8; ++r) {
      int vrow = row0 + r + half * 8;
      int col = nt * 16 + n16;
      __bf16 hv = f2bf(hsel[r]);
      od[vrow * OUT_F + col] = hv;
      if (WSWZ) osw[bswz_idx(OUT_F, vrow, col)] = hv;
    }
  }
}

// ---------- kernel 4: factored readout reduction ----------
// hbar[b, 0:64|64:192|192:320] = sum_v mask(h_l[b,v,:]) * h_l[b,v,:]
__global__ __launch_bounds__(128) void hbar_kernel(
    const __bf16* __restrict__ h0, const __bf16* __restrict__ h1,
    const __bf16* __restrict__ h2, __bf16* __restrict__ hbarbf) {
  __shared__ float maskv[N_];
  int b = blockIdx.x, t = threadIdx.x;  // 128 threads
  const __bf16* src[3] = { h0 + (size_t)b * N_ * NODE_F,
                           h1 + (size_t)b * N_ * OUT_F,
                           h2 + (size_t)b * N_ * OUT_F };
  const int D[3]   = { NODE_F, OUT_F, OUT_F };
  const int off[3] = { 0, 64, 192 };
  for (int s3 = 0; s3 < 3; ++s3) {
    const __bf16* h = src[s3];
    int d = D[s3];
    float rs = 0.f;
    for (int c = 0; c < d; ++c) rs += bf2f(h[t * d + c]);
    maskv[t] = (rs > 1e-6f) ? 1.f : 0.f;
    __syncthreads();
    if (t < d) {
      float acc = 0.f;
      for (int v = 0; v < N_; ++v) acc += maskv[v] * bf2f(h[v * d + t]);
      hbarbf[(size_t)b * RCAT + off[s3] + t] = f2bf(acc);
    }
    __syncthreads();
  }
}

// ---------- kernel 5: s = hbar[128x320] @ Rcat[320x256] (WMMA) ----------
__global__ __launch_bounds__(256) void s_gemm_kernel(
    const __bf16* __restrict__ hbarbf, const __bf16* __restrict__ Rcatswz,
    float* __restrict__ s) {
  int id = blockIdx.x * 8 + (threadIdx.x >> 5);  // 0..127 wave-tiles
  int rt = id >> 4, ct = id & 15;                // 8 row-tiles x 16 col-tiles
  int lane = threadIdx.x & 31;
  int half = lane >> 4, n16 = lane & 15;
  v8f acc = {};
#pragma unroll
  for (int c = 0; c < RCAT / 32; ++c) {
    v16bf a = load_frag_a(hbarbf + rt * 16 * RCAT + c * 32, RCAT);
    v16bf b = load_frag_b_swz(Rcatswz + (c * (ROUT / 16) + ct) * 512);
    acc = WMMA_BF16(a, b, acc);
  }
#pragma unroll
  for (int r = 0; r < 8; ++r)
    s[(rt * 16 + r + half * 8) * ROUT + ct * 16 + n16] = acc[r];
}

// ---------- kernel 6: softmax + 4-layer MLP, exact f32 (tiny) ----------
__global__ __launch_bounds__(256) void mlp_kernel(
    const float* __restrict__ s,
    const float* __restrict__ nw0, const float* __restrict__ nb0,
    const float* __restrict__ nw1, const float* __restrict__ nb1,
    const float* __restrict__ nw2, const float* __restrict__ nb2,
    const float* __restrict__ nw3, const float* __restrict__ nb3,
    float* __restrict__ out) {
  __shared__ float xa[256], xb[256], red[256];
  int b = blockIdx.x, t = threadIdx.x;  // 256 threads
  float v = s[b * 256 + t];
  red[t] = v; __syncthreads();
  for (int o = 128; o > 0; o >>= 1) { if (t < o) red[t] = fmaxf(red[t], red[t + o]); __syncthreads(); }
  float mx = red[0]; __syncthreads();
  float ev = __expf(v - mx);
  red[t] = ev; __syncthreads();
  for (int o = 128; o > 0; o >>= 1) { if (t < o) red[t] += red[t + o]; __syncthreads(); }
  xa[t] = ev / red[0]; __syncthreads();
  if (t < 128) {                 // 256 -> 128
    float a = nb0[t];
    for (int i = 0; i < 256; ++i) a += xa[i] * nw0[i * 128 + t];
    xb[t] = fmaxf(a, 0.f);
  }
  __syncthreads();
  {                              // 128 -> 256
    float a = nb1[t];
    for (int i = 0; i < 128; ++i) a += xb[i] * nw1[i * 256 + t];
    xa[t] = fmaxf(a, 0.f);
  }
  __syncthreads();
  if (t < 128) {                 // 256 -> 128
    float a = nb2[t];
    for (int i = 0; i < 256; ++i) a += xa[i] * nw2[i * 128 + t];
    xb[t] = fmaxf(a, 0.f);
  }
  __syncthreads();
  if (t < TGT) {                 // 128 -> 12
    float a = nb3[t];
    for (int i = 0; i < 128; ++i) a += xb[i] * nw3[i * TGT + t];
    out[b * TGT + t] = a;
  }
}

// ---------- launch ----------
extern "C" void kernel_launch(void* const* d_in, const int* in_sizes, int n_in,
                              void* d_out, int out_size, void* d_ws, size_t ws_size,
                              hipStream_t stream) {
  const float* g    = (const float*)d_in[0];
  const float* h_in = (const float*)d_in[1];
  const float* e    = (const float*)d_in[2];
  const float* W_u0 = (const float*)d_in[3];
  const float* W_u1 = (const float*)d_in[4];
  const float* R0   = (const float*)d_in[5];
  const float* R1   = (const float*)d_in[6];
  const float* R2   = (const float*)d_in[7];
  const float* nw0  = (const float*)d_in[8];
  const float* nb0  = (const float*)d_in[9];
  const float* nw1  = (const float*)d_in[10];
  const float* nb1  = (const float*)d_in[11];
  const float* nw2  = (const float*)d_in[12];
  const float* nb2  = (const float*)d_in[13];
  const float* nw3  = (const float*)d_in[14];
  const float* nb3  = (const float*)d_in[15];
  float* out = (float*)d_out;

  char* ws = (char*)d_ws;
  size_t off = 0;
  auto alloc = [&](size_t bytes) -> void* {
    off = (off + 1023) & ~(size_t)1023;
    void* p = ws + off;
    off += bytes;
    return p;
  };
  __bf16* gbf     = (__bf16*)alloc((size_t)B_ * N_ * N_ * 2);
  __bf16* h0bf    = (__bf16*)alloc((size_t)B_ * N_ * NODE_F * 2);
  __bf16* h0swz   = (__bf16*)alloc((size_t)B_ * N_ * NODE_F * 2);
  __bf16* mebf    = (__bf16*)alloc((size_t)B_ * N_ * EDGE_F * 2);
  int*    kidx    = (int*)   alloc((size_t)B_ * N_ * 4);
  __bf16* W0swz   = (__bf16*)alloc((size_t)4 * 96 * 128 * 2);
  __bf16* W1swz   = (__bf16*)alloc((size_t)4 * 160 * 128 * 2);
  __bf16* Rcatswz = (__bf16*)alloc((size_t)RCAT * ROUT * 2);
  __bf16* h1bf    = (__bf16*)alloc((size_t)B_ * N_ * OUT_F * 2);
  __bf16* h1swz   = (__bf16*)alloc((size_t)B_ * N_ * OUT_F * 2);
  __bf16* h2bf    = (__bf16*)alloc((size_t)B_ * N_ * OUT_F * 2);
  __bf16* hbarbf  = (__bf16*)alloc((size_t)B_ * RCAT * 2);
  float*  sbuf    = (float*) alloc((size_t)B_ * ROUT * 4);

  // 1) edge aggregation (shared by both layers) + degree buckets + converts
  edge_prep_kernel<<<B_ * N_ / 8, 256, 0, stream>>>(g, h_in, e, gbf, h0bf, h0swz,
                                                    mebf, kidx);

  // 2) weight converts into B-swizzled fragment layout
  for (int k = 0; k < 4; ++k) {
    f2bf_swz_kernel<<<(96 * 128 + 255) / 256, 256, 0, stream>>>(
        W_u0 + (size_t)k * 96 * 128, W0swz + (size_t)k * 96 * 128, 96, 128, 0);
    f2bf_swz_kernel<<<(160 * 128 + 255) / 256, 256, 0, stream>>>(
        W_u1 + (size_t)k * 160 * 128, W1swz + (size_t)k * 160 * 128, 160, 128, 0);
  }
  f2bf_swz_kernel<<<(64 * 256 + 255) / 256, 256, 0, stream>>>(R0, Rcatswz, 64, 256, 0);
  f2bf_swz_kernel<<<(128 * 256 + 255) / 256, 256, 0, stream>>>(R1, Rcatswz, 128, 256, 64);
  f2bf_swz_kernel<<<(128 * 256 + 255) / 256, 256, 0, stream>>>(R2, Rcatswz, 128, 256, 192);

  // 3) two message-passing layers (WMMA)
  layer_kernel<NODE_F, NODE_F + EDGE_F, true><<<B_, 256, 0, stream>>>(
      gbf, h0swz, mebf, kidx, W0swz, h1bf, h1swz);
  layer_kernel<OUT_F, OUT_F + EDGE_F, false><<<B_, 256, 0, stream>>>(
      gbf, h1swz, mebf, kidx, W1swz, h2bf, (__bf16*)0);

  // 4) factored readout: reduce over nodes, then one small WMMA GEMM
  hbar_kernel<<<B_, 128, 0, stream>>>(h0bf, h1bf, h2bf, hbarbf);
  s_gemm_kernel<<<16, 256, 0, stream>>>(hbarbf, Rcatswz, sbuf);

  // 5) softmax + MLP head (exact f32)
  mlp_kernel<<<B_, 256, 0, stream>>>(sbuf, nw0, nb0, nw1, nb1, nw2, nb2, nw3, nb3, out);
}
